// Encoder_82738249990729
// MI455X (gfx1250) — compile-verified, matrix operand order checked by reading, not run on hardware
//
#include <hip/hip_runtime.h>
#include <hip/hip_bf16.h>

typedef __attribute__((ext_vector_type(16))) __bf16 v16bf;
typedef __attribute__((ext_vector_type(8)))  __bf16 v8bf;
typedef __attribute__((ext_vector_type(4)))  __bf16 v4bf;
typedef __attribute__((ext_vector_type(8)))  float  v8f;
typedef __attribute__((ext_vector_type(4)))  unsigned int u32x4;
typedef __attribute__((ext_vector_type(8)))  int    i32x8;
typedef __attribute__((ext_vector_type(4)))  int    i32x4;

union FragBF { v16bf v; v8bf h[2]; };

#define B_   64
#define T_   512
#define D_   256
#define H_   512
#define NG   2048   // 4*H gate columns
#define KC   768    // D + H (fused K)
#define NBLK 16     // persistent blocks
#define NSLC 128    // NG / NBLK gate cols per block
#define HSLC 32     // H  / NBLK hidden units per block
#define CTX_OFF 49152            // 64*768 context floats, then traj

// LDS partition byte offsets (dynamic shared memory)
#define LDS_WU   0                         // 128*768 bf16 = 196608
#define LDS_XS   196608                    // 64*256 bf16  =  32768 (aliased by zbuf)
#define LDS_HS   229376                    // 64*512 bf16  =  65536
#define LDS_ZB   196608                    // 64*128 f32   =  32768 (alias of XS)
#define LDS_CB   294912                    // 64*32  f32   =   8192
#define LDS_TOTAL 303104

// ---------------- CDNA5 hardware tanh (transcendental unit) ----------------
__device__ inline float tanh_fast(float x) {
  float r;
  asm("v_tanh_f32 %0, %1" : "=v"(r) : "v"(x));
  return r;
}

// ---------------- Tensor Data Mover: 2D bf16 tile Global -> LDS ------------
// D# per CDNA5 ISA: group0 {count=1, lds_addr,
// global_addr, type=2}, group1 {data_size=2B, tensor_dim0/1, tile_dim0/1,
// 48-bit tensor_dim0_stride}. Groups 2/3 zero (2D tensor).
// This toolchain exposes the 6-arg builtin: (g0, g1, g2, g3, i32x8, cpol).
__device__ inline void tdm_load_2d(unsigned lds_byte_off, const void* gptr,
                                   unsigned dim0, unsigned dim1,
                                   unsigned tile0, unsigned tile1,
                                   unsigned long long stride0_elems) {
  unsigned long long ga = (unsigned long long)(uintptr_t)gptr;
  u32x4 g0;
  g0[0] = 1u;                                              // count=1 (valid)
  g0[1] = lds_byte_off;                                    // lds_addr
  g0[2] = (unsigned)(ga & 0xFFFFFFFFu);                    // global_addr[31:0]
  g0[3] = (unsigned)((ga >> 32) & 0x1FFFFFFu) | (2u << 30);// addr[56:32] | type=2
  i32x8 g1;
  g1[0] = (int)(1u << 16);                                 // data_size=1 -> 2 bytes
  g1[1] = (int)((dim0 & 0xFFFFu) << 16);                   // tensor_dim0[15:0] @bit48
  g1[2] = (int)((dim0 >> 16) | ((dim1 & 0xFFFFu) << 16));  // dim0 hi | dim1 lo
  g1[3] = (int)((dim1 >> 16) | ((tile0 & 0xFFFFu) << 16)); // dim1 hi | tile_dim0
  g1[4] = (int)(tile1 & 0xFFFFu);                          // tile_dim1, tile_dim2=0
  g1[5] = (int)(unsigned)(stride0_elems & 0xFFFFFFFFull);  // dim0_stride[31:0]
  g1[6] = (int)(unsigned)((stride0_elems >> 32) & 0xFFFFull); // stride hi16
  g1[7] = 0;
  i32x4 g2 = {0, 0, 0, 0};
  i32x4 g3 = {0, 0, 0, 0};
  i32x8 g4 = {0, 0, 0, 0, 0, 0, 0, 0};
  __builtin_amdgcn_tensor_load_to_lds(g0, g1, g2, g3, g4, 0);
}

// ---------------- device-wide generation barrier (16 resident blocks) ------
__device__ inline void grid_barrier(unsigned* bar, unsigned nb) {
  __syncthreads();
  if (threadIdx.x == 0) {
    __threadfence();
    unsigned* cnt = bar;
    unsigned* gen = bar + 1;
    unsigned g = __hip_atomic_load(gen, __ATOMIC_ACQUIRE, __HIP_MEMORY_SCOPE_AGENT);
    unsigned v = __hip_atomic_fetch_add(cnt, 1u, __ATOMIC_ACQ_REL, __HIP_MEMORY_SCOPE_AGENT);
    if (v == nb - 1u) {
      __hip_atomic_store(cnt, 0u, __ATOMIC_RELAXED, __HIP_MEMORY_SCOPE_AGENT);
      __hip_atomic_fetch_add(gen, 1u, __ATOMIC_RELEASE, __HIP_MEMORY_SCOPE_AGENT);
    } else {
      while (__hip_atomic_load(gen, __ATOMIC_ACQUIRE, __HIP_MEMORY_SCOPE_AGENT) == g)
        __builtin_amdgcn_s_sleep(1);
    }
  }
  __syncthreads();
}

// ---------------- init: zero h0 (bf16 buffer 0) + barrier words ------------
__global__ void k_init(unsigned* hzero, unsigned* bar) {
  int i = blockIdx.x * blockDim.x + threadIdx.x;
  if (i < (B_ * H_ * 2) / 4) hzero[i] = 0u;   // 64KB = 16384 dwords
  if (i < 2) bar[i] = 0u;
}

// ---------------- x: f32 -> bf16 (4 elems / thread) ------------------------
__global__ void k_cvt_x(const float* __restrict__ x, __bf16* __restrict__ xb, int n4) {
  int i = blockIdx.x * blockDim.x + threadIdx.x;
  if (i >= n4) return;
  float4 v = ((const float4*)x)[i];
  v4bf o;
  o[0] = (__bf16)v.x; o[1] = (__bf16)v.y; o[2] = (__bf16)v.z; o[3] = (__bf16)v.w;
  ((v4bf*)xb)[i] = o;
}

// ---- build permuted [W;U]^T as bf16 [2048 x 768] row-major + permuted bias -
// row rr = w*128 + g*32 + jj  <->  original gate column g*H + (w*32 + jj)
__global__ void k_build_wu(const float* __restrict__ W, const float* __restrict__ U,
                           const float* __restrict__ b, __bf16* __restrict__ wu,
                           float* __restrict__ bperm) {
  int i = blockIdx.x * blockDim.x + threadIdx.x;
  if (i >= NG * KC) return;
  int rr = i / KC, k = i % KC;
  int w = rr >> 7, g = (rr >> 5) & 3, jj = rr & 31;
  int col = g * H_ + w * HSLC + jj;
  float v = (k < D_) ? W[(size_t)k * NG + col] : U[(size_t)(k - D_) * NG + col];
  wu[i] = (__bf16)v;
  if (k == 0) bperm[rr] = b[col];
}

// ---------------- copy x into traj_h[..., 0:256] and context x part --------
__global__ void k_copy_x(const float* __restrict__ x, float* __restrict__ out) {
  int i = blockIdx.x * blockDim.x + threadIdx.x;      // over 64*512*64 float4s
  if (i >= B_ * T_ * (D_ / 4)) return;
  int bidx = i >> 15;                  // / (512*64)
  int rem  = i & 32767;
  int t    = rem >> 6;
  int d4   = rem & 63;
  float4 v = ((const float4*)x)[i];
  *(float4*)(out + CTX_OFF + ((size_t)(bidx * T_ + t)) * 768 + d4 * 4) = v;
  if (t == T_ - 1)
    *(float4*)(out + (size_t)bidx * 768 + d4 * 4) = v;
}

// ---------------- persistent fused LSTM: z = [x_t | h] @ WUslice + b --------
__global__ __launch_bounds__(256, 1)
void k_lstm(const __bf16* __restrict__ xb, const __bf16* __restrict__ wu,
            const float* __restrict__ bperm, __bf16* __restrict__ hg,
            unsigned* __restrict__ bar, float* __restrict__ out) {
  extern __shared__ __align__(16) char smem[];
  __bf16* wu_s = (__bf16*)(smem + LDS_WU);   // 128x768 U/W slice (reused 512x)
  __bf16* x_s  = (__bf16*)(smem + LDS_XS);   // 64x256 x_t tile (TDM-staged)
  __bf16* h_s  = (__bf16*)(smem + LDS_HS);   // 64x512 h tile   (TDM-staged)
  float*  zbuf = (float*)(smem + LDS_ZB);    // 64x128 gates (aliases x_s)
  float*  cbuf = (float*)(smem + LDS_CB);    // 64x32 cell state

  const int w    = blockIdx.x;
  const int tid  = threadIdx.x;
  const int lane = tid & 31;
  const int wave = tid >> 5;          // 8 waves = 8 N-tiles of 16
  const int r    = lane & 15;
  const int kb   = (lane >> 4) << 3;  // 0 or 8: K sub-chunk per WMMA layout
  const unsigned lds_base = (unsigned)__builtin_amdgcn_groupstaticsize();

  // Stage this block's 128x768 bf16 [W;U]^T slice into LDS.
  {
    const uint4* src = (const uint4*)(wu + (size_t)w * NSLC * KC);
    uint4* dst = (uint4*)wu_s;
    for (int i = tid; i < NSLC * KC / 8; i += 256) dst[i] = src[i];
  }
  for (int i = tid; i < B_ * HSLC; i += 256) cbuf[i] = 0.f;  // c0 = 0
  __syncthreads();

  const float bias = bperm[w * NSLC + wave * 16 + r];

  for (int t = 0; t < T_; ++t) {
    const __bf16* hr = hg + (size_t)(t & 1) * (B_ * H_);        // read  h(t-1)
    __bf16*       hw = hg + (size_t)((t + 1) & 1) * (B_ * H_);  // write h(t)

    // ---- TDM: stage x_t (strided 2D tile) and h(t-1) (dense tile) to LDS --
    if (wave == 0) {
      // x_t: 64 rows of 256 bf16, row stride T*D elements, tile start x[0][t][0]
      tdm_load_2d(lds_base + LDS_XS, xb + (size_t)t * D_,
                  (unsigned)(T_ * D_), B_, D_, B_, (unsigned long long)(T_ * D_));
      // h(t-1): dense 64x512 bf16
      tdm_load_2d(lds_base + LDS_HS, hr, H_, B_, H_, B_, H_);
      __builtin_amdgcn_s_wait_tensorcnt(0);
    }
    __syncthreads();

    v8f acc[4] = {};  // 4 M-tiles (batch 0..63) x this wave's 16 gate cols

    // K over x_t (D = 256), A from LDS
    for (int kk = 0; kk < D_; kk += 32) {
      FragBF bfr;
      const __bf16* bp = wu_s + (size_t)(wave * 16 + r) * KC + kk + kb;
      bfr.h[0] = *(const v8bf*)bp;
      bfr.h[1] = *(const v8bf*)(bp + 16);
#pragma unroll
      for (int m = 0; m < 4; ++m) {
        const __bf16* ap = x_s + (size_t)(m * 16 + r) * D_ + kk + kb;
        FragBF afr;
        afr.h[0] = *(const v8bf*)ap;
        afr.h[1] = *(const v8bf*)(ap + 16);
        acc[m] = __builtin_amdgcn_wmma_f32_16x16x32_bf16(
            false, afr.v, false, bfr.v, (short)0, acc[m], false, false);
      }
    }
    // K over h(t-1) (H = 512), A from LDS
    for (int kk = 0; kk < H_; kk += 32) {
      FragBF bfr;
      const __bf16* bp = wu_s + (size_t)(wave * 16 + r) * KC + D_ + kk + kb;
      bfr.h[0] = *(const v8bf*)bp;
      bfr.h[1] = *(const v8bf*)(bp + 16);
#pragma unroll
      for (int m = 0; m < 4; ++m) {
        const __bf16* ap = h_s + (size_t)(m * 16 + r) * H_ + kk + kb;
        FragBF afr;
        afr.h[0] = *(const v8bf*)ap;
        afr.h[1] = *(const v8bf*)(ap + 16);
        acc[m] = __builtin_amdgcn_wmma_f32_16x16x32_bf16(
            false, afr.v, false, bfr.v, (short)0, acc[m], false, false);
      }
    }
    __syncthreads();  // all waves done reading x_s before zbuf (alias) writes

    // Epilogue: z + bias -> LDS (C layout: VGPR i -> row i + (lane>=16)*8)
#pragma unroll
    for (int m = 0; m < 4; ++m)
#pragma unroll
      for (int i = 0; i < 8; ++i) {
        int rowM = m * 16 + i + kb;
        zbuf[rowM * NSLC + wave * 16 + r] = acc[m][i] + bias;
      }
    __syncthreads();

    // Cell update for this block's 32 hidden units x 64 batches.
    for (int itr = 0; itr < 8; ++itr) {
      int idx = tid + itr * 256;        // 0..2047
      int bi = idx >> 5, jj = idx & 31;
      float zi = zbuf[bi * NSLC +       jj];
      float zf = zbuf[bi * NSLC +  32 + jj];
      float zg = zbuf[bi * NSLC +  64 + jj];
      float zo = zbuf[bi * NSLC +  96 + jj];
      float c2 = tanh_fast(zf) * cbuf[idx] + tanh_fast(zi) * tanh_fast(zg);
      cbuf[idx] = c2;
      float h2 = tanh_fast(zo) * tanh_fast(c2);
      int jg = w * HSLC + jj;
      hw[bi * H_ + jg] = (__bf16)h2;
      out[CTX_OFF + ((size_t)(bi * T_) + t) * 768 + 256 + jg] = h2;
      if (t == T_ - 1) out[(size_t)bi * 768 + 256 + jg] = h2;
    }
    grid_barrier(bar, NBLK);  // publish h(t) to all blocks before t+1
  }
}

extern "C" void kernel_launch(void* const* d_in, const int* in_sizes, int n_in,
                              void* d_out, int out_size, void* d_ws, size_t ws_size,
                              hipStream_t stream) {
  (void)in_sizes; (void)n_in; (void)out_size; (void)ws_size;
  const float* x = (const float*)d_in[0];
  const float* W = (const float*)d_in[1];
  const float* U = (const float*)d_in[2];
  const float* b = (const float*)d_in[3];
  float* out = (float*)d_out;

  char* ws = (char*)d_ws;
  __bf16*   xb    = (__bf16*)ws;                       // 64*512*256 bf16 = 16 MB
  __bf16*   wu    = (__bf16*)(ws + 16777216);          // 2048*768 bf16   = 3 MB
  float*    bperm = (float*) (ws + 19922944);          // 2048 f32
  __bf16*   hg    = (__bf16*)(ws + 19931136);          // 2 x 64*512 bf16 (double buffer)
  unsigned* bar   = (unsigned*)(ws + 20062208);        // barrier {cnt, gen}

  k_init   <<<64,   256, 0, stream>>>((unsigned*)hg, bar);
  k_cvt_x  <<<8192, 256, 0, stream>>>(x, xb, B_ * T_ * D_ / 4);
  k_build_wu<<<6144,256, 0, stream>>>(W, U, b, wu, bperm);
  k_copy_x <<<8192, 256, 0, stream>>>(x, out);

  k_lstm<<<NBLK, 256, LDS_TOTAL, stream>>>(xb, wu, bperm, hg, bar, out);
}
